// ContrastiveModel_50706383897166
// MI455X (gfx1250) — matrix operand order repaired and checked by reading, not compiled, not run on hardware
//
#include <hip/hip_runtime.h>
#include <hip/hip_bf16.h>

// ---------------------------------------------------------------------------
// ColBERT-style MaxSim on MI455X (gfx1250, wave32, WMMA).
//   pass1: fp32 -> row-normalize -> f16 (workspace)
//   pass2: tiled GEMM with v_wmma_f32_16x16x32_f16, fused masked row/col max
//          A strip staged to LDS with GLOBAL_LOAD_ASYNC_TO_LDS_B128
//   pass3: per-batch masked sums -> score
// ---------------------------------------------------------------------------

typedef __attribute__((ext_vector_type(16))) _Float16 v16h;
typedef __attribute__((ext_vector_type(8)))  float    v8f;
typedef __attribute__((ext_vector_type(4)))  unsigned v4u;

static constexpr int kB = 64;
static constexpr int kS = 512;
static constexpr int kD = 768;
static constexpr int kLDA = 776;          // 768 + 8 halves pad -> conflict-free ds_load_b128
static constexpr float kNEG = -1e9f;

// order-preserving float <-> uint mapping (for atomicMax over signed floats)
__device__ __forceinline__ unsigned f2mono(float f) {
  unsigned u = __float_as_uint(f);
  return (u & 0x80000000u) ? ~u : (u | 0x80000000u);
}
__device__ __forceinline__ float mono2f(unsigned u) {
  return __uint_as_float((u & 0x80000000u) ? (u & 0x7fffffffu) : ~u);
}

// --------------------------- pass 0a: prefix counts ------------------------
__global__ __launch_bounds__(256) void k_counts(const int* __restrict__ m1,
                                                const int* __restrict__ m2,
                                                int* __restrict__ n1,
                                                int* __restrict__ n2) {
  __shared__ int sh[2];
  int b = blockIdx.x, t = threadIdx.x;
  if (t == 0) { sh[0] = 0; sh[1] = 0; }
  __syncthreads();
  int a = 0, c = 0;
  for (int i = t; i < kS; i += 256) { a += m1[b * kS + i]; c += m2[b * kS + i]; }
  for (int m = 16; m >= 1; m >>= 1) { a += __shfl_xor(a, m, 32); c += __shfl_xor(c, m, 32); }
  if ((t & 31) == 0) { atomicAdd(&sh[0], a); atomicAdd(&sh[1], c); }
  __syncthreads();
  if (t == 0) { n1[b] = sh[0]; n2[b] = sh[1]; }
}

// --------------------------- pass 0b: init colmax --------------------------
__global__ __launch_bounds__(256) void k_init_colmax(unsigned* __restrict__ cm) {
  int i = blockIdx.x * 256 + threadIdx.x;
  if (i < kB * kS) cm[i] = f2mono(kNEG);
}

// ------------------- pass 1: row-normalize fp32 -> f16 ---------------------
// one wave (32 lanes) per row of 768 floats
__global__ __launch_bounds__(256) void k_normcvt(const float* __restrict__ src,
                                                 _Float16* __restrict__ dst) {
  int row  = blockIdx.x * 8 + (threadIdx.x >> 5);
  int lane = threadIdx.x & 31;
  const float* r = src + (size_t)row * kD;
  float vals[24];
  float ss = 0.f;
#pragma unroll
  for (int t = 0; t < 24; ++t) { float v = r[lane + t * 32]; vals[t] = v; ss += v * v; }
#pragma unroll
  for (int m = 16; m >= 1; m >>= 1) ss += __shfl_xor(ss, m, 32);
  float scale = 1.0f / fmaxf(sqrtf(ss), 1e-4f);   // eps guard (never binds for this data)
  _Float16* o = dst + (size_t)row * kD;
#pragma unroll
  for (int t = 0; t < 24; ++t) o[lane + t * 32] = (_Float16)(vals[t] * scale);
}

// ------------- pass 2: WMMA GEMM strip + fused masked max-reduce -----------
// grid: (16 strips of 32 rows, 64 batches); 256 threads = 8 waves
// wave tile: 32 rows x 64 cols -> all 512 cols in one pass (8 waves)
__global__ __launch_bounds__(256) void k_gemm_maxsim(
    const _Float16* __restrict__ a1h, const _Float16* __restrict__ a2h,
    const int* __restrict__ n1v, const int* __restrict__ n2v,
    float* __restrict__ rowmax, unsigned* __restrict__ colm) {
  __shared__ _Float16 As[32 * kLDA];     // staged A strip (f16, padded rows)
  __shared__ unsigned rowsh[32];         // per-strip rowmax (monotonic uint)

  const int b     = blockIdx.y;
  const int i0    = blockIdx.x * 32;
  const int n1    = n1v[b];
  const int n2    = n2v[b];
  const _Float16* A1 = a1h + ((size_t)b * kS + i0) * kD;
  const _Float16* A2 = a2h + (size_t)b * kS * kD;

  // --- stage 32x768 halves -> LDS via async global->LDS DMA (16B per lane) ---
  // 32*768 halves = 3072 16B chunks = 256 threads x 12
  {
#pragma unroll
    for (int it = 0; it < 12; ++it) {
      int t = threadIdx.x + it * 256;
      int r = t / 96, c = t % 96;                         // 96 chunks per row
      const _Float16* g = A1 + (size_t)r * kD + c * 8;    // global source
      unsigned lo = (unsigned)(size_t)(const void*)(As + r * kLDA + c * 8);
      asm volatile("global_load_async_to_lds_b128 %0, %1, off"
                   :: "v"(lo), "v"(g) : "memory");
    }
    if (threadIdx.x < 32) rowsh[threadIdx.x] = f2mono(kNEG);
    asm volatile("s_wait_asynccnt 0x0" ::: "memory");     // drain this wave's DMAs
  }
  __syncthreads();

  const int wave = threadIdx.x >> 5;     // 0..7 -> 8 N-groups of 64 cols
  const int lane = threadIdx.x & 31;
  const int lrow = lane & 15;
  const int khi  = lane >> 4;            // 0/1
  const int j0   = wave * 64;

  float rm[2][8];
#pragma unroll
  for (int m = 0; m < 2; ++m)
#pragma unroll
    for (int v = 0; v < 8; ++v) rm[m][v] = kNEG;

  union HF { v16h h; v4u u[2]; };

  v8f acc[2][4] = {};                    // [msub][nsub]

  const _Float16* ap0 = As + (0 * 16 + lrow) * kLDA + khi * 8;
  const _Float16* ap1 = As + (1 * 16 + lrow) * kLDA + khi * 8;
  const _Float16* bp[4];
#pragma unroll
  for (int n = 0; n < 4; ++n)
    bp[n] = A2 + (size_t)(j0 + n * 16 + lrow) * kD + khi * 8;

#pragma unroll 4
  for (int k0 = 0; k0 < kD; k0 += 32) {
    HF a0; a0.u[0] = *(const v4u*)(ap0 + k0); a0.u[1] = *(const v4u*)(ap0 + k0 + 16);
    HF a1; a1.u[0] = *(const v4u*)(ap1 + k0); a1.u[1] = *(const v4u*)(ap1 + k0 + 16);
    HF bf[4];
#pragma unroll
    for (int n = 0; n < 4; ++n) {
      bf[n].u[0] = *(const v4u*)(bp[n] + k0);
      bf[n].u[1] = *(const v4u*)(bp[n] + k0 + 16);
    }
#pragma unroll
    for (int n = 0; n < 4; ++n) {
      acc[0][n] = __builtin_amdgcn_wmma_f32_16x16x32_f16(false, a0.h, false, bf[n].h,
                                                         (short)0, acc[0][n], false, false);
      acc[1][n] = __builtin_amdgcn_wmma_f32_16x16x32_f16(false, a1.h, false, bf[n].h,
                                                         (short)0, acc[1][n], false, false);
    }
  }

  // epilogue: C/D layout -> j = jtile + (lane&15); i = itile + khi*8 + vgpr
#pragma unroll
  for (int n = 0; n < 4; ++n) {
    const int j  = j0 + n * 16 + lrow;
    const bool vj = j < n2;
    float cmx = kNEG;
#pragma unroll
    for (int m = 0; m < 2; ++m) {
#pragma unroll
      for (int v = 0; v < 8; ++v) {
        const int i = i0 + m * 16 + khi * 8 + v;
        const float val = (vj && (i < n1)) ? acc[m][n][v] : kNEG;
        rm[m][v] = fmaxf(rm[m][v], val);
        cmx = fmaxf(cmx, val);
      }
    }
    cmx = fmaxf(cmx, __shfl_xor(cmx, 16, 32));       // fold other 16 rows (same j)
    if (lane < 16) atomicMax(&colm[(size_t)b * kS + j], f2mono(cmx));
  }

  // rowmax: fold across the 16 lanes sharing each row, then ds-atomicMax
#pragma unroll
  for (int m = 0; m < 2; ++m)
#pragma unroll
    for (int v = 0; v < 8; ++v) {
      float x = rm[m][v];
#pragma unroll
      for (int msk = 8; msk >= 1; msk >>= 1) x = fmaxf(x, __shfl_xor(x, msk, 32));
      rm[m][v] = x;
    }
  if (lrow == 0) {
#pragma unroll
    for (int m = 0; m < 2; ++m)
#pragma unroll
      for (int v = 0; v < 8; ++v)
        atomicMax(&rowsh[m * 16 + khi * 8 + v], f2mono(rm[m][v]));
  }
  __syncthreads();
  if (threadIdx.x < 32)
    rowmax[(size_t)b * kS + i0 + threadIdx.x] = mono2f(rowsh[threadIdx.x]);
}

// ------------------------- pass 3: per-batch score -------------------------
__global__ __launch_bounds__(256) void k_finalize(const float* __restrict__ rowmax,
                                                  const unsigned* __restrict__ colm,
                                                  const int* __restrict__ n1v,
                                                  const int* __restrict__ n2v,
                                                  float* __restrict__ out) {
  __shared__ float sh[8];
  const int b = blockIdx.x, t = threadIdx.x;
  const int n1 = n1v[b], n2 = n2v[b];
  float part = 0.f;
  for (int i = t; i < kS; i += 256) {
    if (i < n1) part += rowmax[(size_t)b * kS + i];
    if (i < n2) part += mono2f(colm[(size_t)b * kS + i]);
  }
  for (int m = 16; m >= 1; m >>= 1) part += __shfl_xor(part, m, 32);
  if ((t & 31) == 0) sh[t >> 5] = part;
  __syncthreads();
  if (t == 0) {
    float tot = 0.f;
    for (int w = 0; w < 8; ++w) tot += sh[w];
    out[b] = tot / (float)(n1 + n2);
  }
}

// ---------------------------------------------------------------------------
extern "C" void kernel_launch(void* const* d_in, const int* in_sizes, int n_in,
                              void* d_out, int out_size, void* d_ws, size_t ws_size,
                              hipStream_t stream) {
  (void)in_sizes; (void)n_in; (void)out_size; (void)ws_size;

  const float* a1 = (const float*)d_in[0];
  const float* a2 = (const float*)d_in[1];
  const int*   m1 = (const int*)d_in[2];
  const int*   m2 = (const int*)d_in[3];

  char* ws = (char*)d_ws;
  const size_t HALF_TENSOR = (size_t)kB * kS * kD * sizeof(_Float16);  // 48 MB
  _Float16* a1h   = (_Float16*)(ws);
  _Float16* a2h   = (_Float16*)(ws + HALF_TENSOR);
  float*    rmax  = (float*)   (ws + 2 * HALF_TENSOR);
  unsigned* cmax  = (unsigned*)(ws + 2 * HALF_TENSOR + (size_t)kB * kS * 4);
  int*      n1    = (int*)     (ws + 2 * HALF_TENSOR + (size_t)kB * kS * 8);
  int*      n2    = n1 + 64;

  k_counts<<<kB, 256, 0, stream>>>(m1, m2, n1, n2);
  k_init_colmax<<<(kB * kS) / 256, 256, 0, stream>>>(cmax);
  k_normcvt<<<(kB * kS) / 8, 256, 0, stream>>>(a1, a1h);
  k_normcvt<<<(kB * kS) / 8, 256, 0, stream>>>(a2, a2h);

  dim3 grid(kS / 32, kB);   // 16 strips x 64 batches
  k_gemm_maxsim<<<grid, 256, 0, stream>>>(a1h, a2h, n1, n2, rmax, cmax);

  k_finalize<<<kB, 256, 0, stream>>>(rmax, cmax, n1, n2, (float*)d_out);
}
